// GCNIIForDialog_18923625906417
// MI455X (gfx1250) — matrix operand order repaired and checked by reading, not compiled, not run on hardware
//
#include <hip/hip_runtime.h>
#include <hip/hip_bf16.h>
#include <math.h>

#define N_NODES   50000
#define N_EDGES   1600000
#define D_INF     1024
#define HF        256
#define L_LAYERS  64
#define N_CLASSES 4
#define ALPHA_C   0.2f
#define THETA_C   0.5f
#define NP        50016   // N padded to multiple of 32 (WMMA needs EXEC all-ones)

typedef __attribute__((ext_vector_type(16))) __bf16 v16bf;
typedef __attribute__((ext_vector_type(8)))  float  v8f;

union BF16x16 { uint4 u[2]; v16bf v; };

__device__ __forceinline__ unsigned short f2bf(float f) {
  unsigned u = __float_as_uint(f);
  unsigned r = u + 0x7FFFu + ((u >> 16) & 1u);   // round-to-nearest-even
  return (unsigned short)(r >> 16);
}

// ---------------- graph normalization ----------------
__global__ void k_deg_init(float* deg) {
  int i = blockIdx.x * blockDim.x + threadIdx.x;
  if (i < N_NODES) deg[i] = 1.0f;                 // self-loop
}
__global__ void k_deg_acc(const int* __restrict__ dst, float* __restrict__ deg) {
  int e = blockIdx.x * blockDim.x + threadIdx.x;
  if (e < N_EDGES) atomicAdd(deg + dst[e], 1.0f);
}
__global__ void k_dinv(float* deg) {
  int i = blockIdx.x * blockDim.x + threadIdx.x;
  if (i < N_NODES) { float d = deg[i]; deg[i] = (d > 0.f) ? rsqrtf(d) : 0.f; }
}
__global__ void k_norm(const int* __restrict__ src, const int* __restrict__ dst,
                       const float* __restrict__ dinv, float* __restrict__ norm) {
  int e = blockIdx.x * blockDim.x + threadIdx.x;
  if (e < N_EDGES) norm[e] = dinv[src[e]] * dinv[dst[e]];
}
__global__ void k_nself(const float* __restrict__ dinv, float* __restrict__ nself) {
  int i = blockIdx.x * blockDim.x + threadIdx.x;
  if (i < N_NODES) { float d = dinv[i]; nself[i] = d * d; }
}

// ---------------- bf16 conversions ----------------
__global__ void k_cvt_x(const float* __restrict__ x, unsigned short* __restrict__ xb) {
  size_t i = (size_t)blockIdx.x * blockDim.x + threadIdx.x;
  if (i < (size_t)N_NODES * D_INF) xb[i] = f2bf(x[i]);
}
// pWt[n][k] = proj_W[k][n]  (K-contiguous for WMMA B-fragment loads)
__global__ void k_cvt_pw(const float* __restrict__ pW, unsigned short* __restrict__ pWt) {
  int i = blockIdx.x * blockDim.x + threadIdx.x;   // i = n*D_INF + k
  if (i < HF * D_INF) {
    int n = i / D_INF, k = i % D_INF;
    pWt[i] = f2bf(pW[(size_t)k * HF + n]);
  }
}
// Wt[l][n][k] = W[l][k][n]
__global__ void k_cvt_w(const float* __restrict__ W, unsigned short* __restrict__ Wt) {
  size_t i = (size_t)blockIdx.x * blockDim.x + threadIdx.x;  // l*H*H + n*H + k
  if (i < (size_t)L_LAYERS * HF * HF) {
    size_t l = i / (HF * HF);
    int r = (int)(i % (HF * HF));
    int n = r / HF, k = r % HF;
    Wt[i] = f2bf(W[l * HF * HF + (size_t)k * HF + n]);
  }
}

// ---------------- WMMA core ----------------
// Wave computes a 16x64 band: A fragment reused across 4 B fragments.
// A (16x32 bf16): lane {m=l&15, hi=l>>4}: 16 contiguous K vals @ k0+hi*16.
// B (32x16 bf16): lane {n=l&15, hi}: 16 contiguous K vals @ k0+hi*16 (transposed W).
// D (16x16 f32):  lane {n=l&15, hi}: VGPR r -> row m0 + r + hi*8.
// Software-pipelined: next K-step fragments are loaded before current WMMAs so
// the loads overlap matrix-core execution instead of s_wait_loadcnt 0.
template<int K, int LD>
__device__ __forceinline__ void mma_band(const unsigned short* __restrict__ Abase,
                                         const unsigned short* __restrict__ Bbase,
                                         v8f acc[4]) {
  BF16x16 a, b[4];
  a.u[0] = *(const uint4*)(Abase);
  a.u[1] = *(const uint4*)(Abase + 8);
#pragma unroll
  for (int t = 0; t < 4; ++t) {
    b[t].u[0] = *(const uint4*)(Bbase + (size_t)t * 16 * LD);
    b[t].u[1] = *(const uint4*)(Bbase + (size_t)t * 16 * LD + 8);
  }
#pragma unroll 2
  for (int k0 = 0; k0 < K; k0 += 32) {
    BF16x16 an, bn[4];
    if (k0 + 32 < K) {
      an.u[0] = *(const uint4*)(Abase + k0 + 32);
      an.u[1] = *(const uint4*)(Abase + k0 + 40);
#pragma unroll
      for (int t = 0; t < 4; ++t) {
        bn[t].u[0] = *(const uint4*)(Bbase + (size_t)t * 16 * LD + k0 + 32);
        bn[t].u[1] = *(const uint4*)(Bbase + (size_t)t * 16 * LD + k0 + 40);
      }
    }
#pragma unroll
    for (int t = 0; t < 4; ++t)
      acc[t] = __builtin_amdgcn_wmma_f32_16x16x32_bf16(
          false, a.v, false, b[t].v, (short)0, acc[t], false, false);
    a = an;
#pragma unroll
    for (int t = 0; t < 4; ++t) b[t] = bn[t];
  }
}

__global__ void __launch_bounds__(256)
k_gemm_proj(const unsigned short* __restrict__ xb, const unsigned short* __restrict__ pWt,
            const float* __restrict__ pb, float* __restrict__ h0, float* __restrict__ h) {
  const int lane = threadIdx.x & 31, wave = threadIdx.x >> 5;
  const int m0 = (blockIdx.x * 2 + (wave >> 2)) * 16;
  const int n0 = (wave & 3) * 64;
  const int l15 = lane & 15, hi = lane >> 4;
  v8f acc[4] = {};
  const unsigned short* Abase = xb + (size_t)(m0 + l15) * D_INF + hi * 16;
  const unsigned short* Bbase = pWt + (size_t)(n0 + l15) * D_INF + hi * 16;
  mma_band<D_INF, D_INF>(Abase, Bbase, acc);
#pragma unroll
  for (int t = 0; t < 4; ++t) {
    int n = n0 + t * 16 + l15;
    float bias = pb[n];
#pragma unroll
    for (int r = 0; r < 8; ++r) {
      size_t m = (size_t)(m0 + r + hi * 8);
      float v = acc[t][r] + bias;
      h0[m * HF + n] = v;
      h [m * HF + n] = v;
    }
  }
}

__global__ void __launch_bounds__(256)
k_gemm_layer(const unsigned short* __restrict__ sb, const float* __restrict__ s,
             const unsigned short* __restrict__ Wt, float* __restrict__ h, float beta) {
  const int lane = threadIdx.x & 31, wave = threadIdx.x >> 5;
  const int m0 = (blockIdx.x * 2 + (wave >> 2)) * 16;
  const int n0 = (wave & 3) * 64;
  const int l15 = lane & 15, hi = lane >> 4;
  v8f acc[4] = {};
  const unsigned short* Abase = sb + (size_t)(m0 + l15) * HF + hi * 16;
  const unsigned short* Bbase = Wt + (size_t)(n0 + l15) * HF + hi * 16;
  mma_band<HF, HF>(Abase, Bbase, acc);
  const float omb = 1.0f - beta;
#pragma unroll
  for (int t = 0; t < 4; ++t) {
    int n = n0 + t * 16 + l15;
#pragma unroll
    for (int r = 0; r < 8; ++r) {
      size_t m = (size_t)(m0 + r + hi * 8);
      float v = omb * s[m * HF + n] + beta * acc[t][r];
      h[m * HF + n] = fmaxf(v, 0.0f);     // ReLU
    }
  }
}

// ---------------- per-layer SpMM pipeline ----------------
// agg[i] := nself[i] * h[i]   (self-loop term; also zeroes pad rows)
__global__ void k_init_agg(const float* __restrict__ h, const float* __restrict__ nself,
                           float* __restrict__ agg) {
  int i = blockIdx.x * blockDim.x + threadIdx.x;      // float4 units
  if (i >= NP * (HF / 4)) return;
  int row = i / (HF / 4);
  float4 v = make_float4(0.f, 0.f, 0.f, 0.f);
  if (row < N_NODES) {
    float w = nself[row];
    float4 hv = ((const float4*)h)[i];
    v = make_float4(w * hv.x, w * hv.y, w * hv.z, w * hv.w);
  }
  ((float4*)agg)[i] = v;
}

// one wave per edge: gather 256 floats of h[src], scatter-add into agg[dst].
// Edge scalars are forced into SGPRs (wave-uniform) so the only vector memory
// traffic is the 2x b128 gather + 8 f32 atomics.
__global__ void __launch_bounds__(256)
k_spmm(const int* __restrict__ src, const int* __restrict__ dst,
       const float* __restrict__ norm, const float* __restrict__ h,
       float* __restrict__ agg) {
  int e = __builtin_amdgcn_readfirstlane((blockIdx.x * blockDim.x + threadIdx.x) >> 5);
  int lane = threadIdx.x & 31;
  int sN = __builtin_amdgcn_readfirstlane(src[e]);
  int dN = __builtin_amdgcn_readfirstlane(dst[e]);
  float w = __uint_as_float(__builtin_amdgcn_readfirstlane(__float_as_uint(norm[e])));
  const float4* hp = (const float4*)(h + (size_t)sN * HF);
  float4 a = hp[lane];
  float4 b = hp[lane + 32];
  float* ap = agg + (size_t)dN * HF;
  int o = lane * 4;
  atomicAdd(ap + o + 0, w * a.x); atomicAdd(ap + o + 1, w * a.y);
  atomicAdd(ap + o + 2, w * a.z); atomicAdd(ap + o + 3, w * a.w);
  o += 128;
  atomicAdd(ap + o + 0, w * b.x); atomicAdd(ap + o + 1, w * b.y);
  atomicAdd(ap + o + 2, w * b.z); atomicAdd(ap + o + 3, w * b.w);
}

// s = (1-alpha)*agg + alpha*h0  (in place into agg); also emit bf16 copy for WMMA A
__global__ void k_combine(float* __restrict__ agg, const float* __restrict__ h0,
                          unsigned short* __restrict__ sb) {
  int i = blockIdx.x * blockDim.x + threadIdx.x;      // float4 units
  if (i >= NP * (HF / 4)) return;
  int row = i / (HF / 4);
  float4 sv = make_float4(0.f, 0.f, 0.f, 0.f);
  if (row < N_NODES) {
    float4 a = ((const float4*)agg)[i];
    float4 z = ((const float4*)h0)[i];
    sv.x = (1.0f - ALPHA_C) * a.x + ALPHA_C * z.x;
    sv.y = (1.0f - ALPHA_C) * a.y + ALPHA_C * z.y;
    sv.z = (1.0f - ALPHA_C) * a.z + ALPHA_C * z.z;
    sv.w = (1.0f - ALPHA_C) * a.w + ALPHA_C * z.w;
  }
  ((float4*)agg)[i] = sv;
  ushort4 p;
  p.x = f2bf(sv.x); p.y = f2bf(sv.y); p.z = f2bf(sv.z); p.w = f2bf(sv.w);
  ((ushort4*)sb)[i] = p;
}

// ---------------- classifier ----------------
__global__ void k_cls(const float* __restrict__ h, const float* __restrict__ cW,
                      const float* __restrict__ cb, float* __restrict__ out) {
  int i = blockIdx.x * blockDim.x + threadIdx.x;
  if (i >= N_NODES * N_CLASSES) return;
  int n = i >> 2, c = i & 3;
  float acc = cb[c];
  const float4* hr = (const float4*)(h + (size_t)n * HF);
#pragma unroll 4
  for (int k = 0; k < HF / 4; ++k) {
    float4 hv = hr[k];
    acc = fmaf(hv.x, cW[(k * 4 + 0) * N_CLASSES + c], acc);
    acc = fmaf(hv.y, cW[(k * 4 + 1) * N_CLASSES + c], acc);
    acc = fmaf(hv.z, cW[(k * 4 + 2) * N_CLASSES + c], acc);
    acc = fmaf(hv.w, cW[(k * 4 + 3) * N_CLASSES + c], acc);
  }
  out[i] = acc;
}

// ---------------- launcher ----------------
extern "C" void kernel_launch(void* const* d_in, const int* in_sizes, int n_in,
                              void* d_out, int out_size, void* d_ws, size_t ws_size,
                              hipStream_t stream) {
  const float* x   = (const float*)d_in[0];
  const int*   src = (const int*)d_in[1];
  const int*   dst = src + N_EDGES;
  const float* pW  = (const float*)d_in[2];
  const float* pb  = (const float*)d_in[3];
  const float* W   = (const float*)d_in[4];
  const float* cW  = (const float*)d_in[5];
  const float* cb  = (const float*)d_in[6];
  float* out = (float*)d_out;

  char* ws = (char*)d_ws;
  size_t off = 0;
  auto alloc = [&](size_t bytes) -> void* {
    void* p = ws + off;
    off += (bytes + 255) & ~(size_t)255;
    return p;
  };
  float*          dinv  = (float*)alloc((size_t)N_NODES * 4);
  float*          norm  = (float*)alloc((size_t)N_EDGES * 4);
  float*          nself = (float*)alloc((size_t)N_NODES * 4);
  unsigned short* xb    = (unsigned short*)alloc((size_t)NP * D_INF * 2);
  unsigned short* pWt   = (unsigned short*)alloc((size_t)HF * D_INF * 2);
  unsigned short* Wt    = (unsigned short*)alloc((size_t)L_LAYERS * HF * HF * 2);
  float*          h0    = (float*)alloc((size_t)NP * HF * 4);
  float*          h     = (float*)alloc((size_t)NP * HF * 4);
  float*          agg   = (float*)alloc((size_t)NP * HF * 4);   // becomes s in-place
  unsigned short* sb    = (unsigned short*)alloc((size_t)NP * HF * 2);

  const int TB = 256;
  const int gN   = (N_NODES + TB - 1) / TB;
  const int gE   = (N_EDGES + TB - 1) / TB;
  const int gNH4 = (NP * (HF / 4) + TB - 1) / TB;

  // graph normalization
  k_deg_init<<<gN, TB, 0, stream>>>(dinv);
  k_deg_acc<<<gE, TB, 0, stream>>>(dst, dinv);
  k_dinv<<<gN, TB, 0, stream>>>(dinv);
  k_norm<<<gE, TB, 0, stream>>>(src, dst, dinv, norm);
  k_nself<<<gN, TB, 0, stream>>>(dinv, nself);

  // bf16 conversions
  k_cvt_x<<<(int)(((size_t)N_NODES * D_INF + TB - 1) / TB), TB, 0, stream>>>(x, xb);
  k_cvt_pw<<<(HF * D_INF + TB - 1) / TB, TB, 0, stream>>>(pW, pWt);
  k_cvt_w<<<(int)(((size_t)L_LAYERS * HF * HF + TB - 1) / TB), TB, 0, stream>>>(W, Wt);

  // h0 = x @ proj_W + b ;  h = h0
  k_gemm_proj<<<NP / 32, TB, 0, stream>>>(xb, pWt, pb, h0, h);

  // 64 GCNII layers
  for (int l = 0; l < L_LAYERS; ++l) {
    float beta = logf(THETA_C / (float)(l + 1) + 1.0f);
    k_init_agg<<<gNH4, TB, 0, stream>>>(h, nself, agg);
    k_spmm<<<N_EDGES / 8, TB, 0, stream>>>(src, dst, norm, h, agg);
    k_combine<<<gNH4, TB, 0, stream>>>(agg, h0, sb);
    k_gemm_layer<<<NP / 32, TB, 0, stream>>>(sb, agg, Wt + (size_t)l * HF * HF, h, beta);
  }

  // classifier
  k_cls<<<(N_NODES * N_CLASSES + TB - 1) / TB, TB, 0, stream>>>(h, cW, cb, out);
}
